// MultiHeadAttention_78718160601547
// MI455X (gfx1250) — matrix operand order repaired and checked by reading, not compiled, not run on hardware
//
#include <hip/hip_runtime.h>

typedef __attribute__((ext_vector_type(16))) _Float16 v16h;
typedef __attribute__((ext_vector_type(8)))  float    v8f;
typedef __attribute__((ext_vector_type(2)))  _Float16 h2;

#define D_MODEL 1024
#define N_HEADS 16
#define D_K     64
#define SEQ     2048
#define BATCH   2
#define ROWS    (BATCH * SEQ)   // 4096

// ---- WMMA fragment loaders (layouts per CDNA5 ISA) ----
// A (16x32 f16): lane holds row m=lane&15; half=lane>>4 selects K interleave:
//   half0 -> K {0..7, 16..23}, half1 -> K {8..15, 24..31}, 2 values per VGPR.
__device__ __forceinline__ v16h load_a_frag(const _Float16* row_base, int kk, int half) {
    v16h a;
#pragma unroll
    for (int v = 0; v < 8; ++v) {
        int k0 = kk + ((v < 4) ? (8 * half + 2 * v) : (16 + 8 * half + 2 * (v - 4)));
        h2 p = *(const h2*)(row_base + k0);
        a[2 * v] = p.x; a[2 * v + 1] = p.y;
    }
    return a;
}

// B (32x16 f16): lane holds column n=lane&15; same K interleave as A: half0 -> K {0..7, 16..23}, half1 -> K {8..15, 24..31}.
__device__ __forceinline__ v16h load_b_frag(const _Float16* col_base, int kk, int half) {
    v16h b;
#pragma unroll
    for (int v = 0; v < 8; ++v) {
        int k0 = kk + ((v < 4) ? (8 * half + 2 * v) : (16 + 8 * half + 2 * (v - 4)));
        h2 p = *(const h2*)(col_base + k0);
        b[2 * v] = p.x; b[2 * v + 1] = p.y;
    }
    return b;
}

// ---- DPP16 16-lane reductions (pure VALU: no ds_bpermute / dscnt waits) ----
// Combining schedule xor1, xor2, xor7, xor15 touches all 16 lanes of a row.
#define DPP_QUAD_PERM_1032  0xB1   // 1 | 0<<2 | 3<<4 | 2<<6
#define DPP_QUAD_PERM_2301  0x4E   // 2 | 3<<2 | 0<<4 | 1<<6
#define DPP_ROW_HALF_MIRROR 0x141
#define DPP_ROW_MIRROR      0x140

template <int CTRL>
__device__ __forceinline__ float dpp_bcast_f32(float x) {
    int t = __builtin_amdgcn_update_dpp(0, __float_as_int(x), CTRL, 0xF, 0xF, true);
    return __int_as_float(t);
}
__device__ __forceinline__ float row16_max(float x) {
    x = fmaxf(x, dpp_bcast_f32<DPP_QUAD_PERM_1032>(x));
    x = fmaxf(x, dpp_bcast_f32<DPP_QUAD_PERM_2301>(x));
    x = fmaxf(x, dpp_bcast_f32<DPP_ROW_HALF_MIRROR>(x));
    x = fmaxf(x, dpp_bcast_f32<DPP_ROW_MIRROR>(x));
    return x;
}
__device__ __forceinline__ float row16_sum(float x) {
    x += dpp_bcast_f32<DPP_QUAD_PERM_1032>(x);
    x += dpp_bcast_f32<DPP_QUAD_PERM_2301>(x);
    x += dpp_bcast_f32<DPP_ROW_HALF_MIRROR>(x);
    x += dpp_bcast_f32<DPP_ROW_MIRROR>(x);
    return x;
}

// ---- fp32 -> fp16 elementwise convert ----
__global__ void cvt_f32_to_f16(const float* __restrict__ src, _Float16* __restrict__ dst, int n) {
    int i = blockIdx.x * blockDim.x + threadIdx.x;
    if (i < n) dst[i] = (_Float16)src[i];
}

// ---- transpose 1024x1024 fp32 weight -> fp16, LDS-tiled 32x32 ----
__global__ void transpose_w_f16(const float* __restrict__ w, _Float16* __restrict__ wt) {
    __shared__ float tile[32][33];
    int tx = threadIdx.x & 31;
    int ty = threadIdx.x >> 5;          // 0..7
    int k0 = blockIdx.x * 32;
    int n0 = blockIdx.y * 32;
#pragma unroll
    for (int i = 0; i < 4; ++i)
        tile[ty + i * 8][tx] = w[(size_t)(k0 + ty + i * 8) * D_MODEL + (n0 + tx)];
    __syncthreads();
#pragma unroll
    for (int i = 0; i < 4; ++i)
        wt[(size_t)(n0 + ty + i * 8) * D_MODEL + (k0 + tx)] = (_Float16)tile[tx][ty + i * 8];
}

// ---- GEMM: [4096x1024] x [1024x1024] (+bias). A f16 row-major, Bt = W^T f16.
// Register-blocked: each wave computes 32x32 (4 WMMA acc), waves tile 4x2 ->
// block tile 128x64. Grid (32, 16), block 256.
// MODE 0: out f16 [B,H,S,64]; MODE 1: out f16 [B,H,64,S]; MODE 2: out f32.
template <int MODE>
__device__ __forceinline__ void store_tile(v8f c, int rbase, int nbase, float bv,
                                           _Float16* __restrict__ out16,
                                           float* __restrict__ outf,
                                           int half, int nl) {
#pragma unroll
    for (int r = 0; r < 8; ++r) {
        const int row = rbase + r + 8 * half;
        const int col = nbase + nl;
        const float val = c[r] + bv;
        if (MODE == 2) {
            outf[(size_t)row * D_MODEL + col] = val;
        } else {
            const int b_ = row >> 11, s = row & 2047;
            const int h  = col >> 6,  d = col & 63;
            if (MODE == 0)
                out16[(((size_t)(b_ * N_HEADS + h)) * SEQ + s) * D_K + d] = (_Float16)val;
            else
                out16[(((size_t)(b_ * N_HEADS + h)) * D_K + d) * SEQ + s] = (_Float16)val;
        }
    }
}

template <int MODE>
__global__ void gemm_f16_wmma(const _Float16* __restrict__ A,
                              const _Float16* __restrict__ Bt,
                              const float* __restrict__ bias,
                              _Float16* __restrict__ out16,
                              float* __restrict__ outf) {
    const int lane = threadIdx.x & 31;
    const int wave = threadIdx.x >> 5;
    const int half = lane >> 4;
    const int nl   = lane & 15;
    const int wr = wave & 3, wc = wave >> 2;       // 4x2 wave grid
    const int r0 = blockIdx.x * 128 + wr * 32;     // rows r0..r0+31
    const int n0 = blockIdx.y * 64  + wc * 32;     // cols n0..n0+31

    const _Float16* arow0 = A  + (size_t)(r0 + nl) * D_MODEL;
    const _Float16* arow1 = arow0 + (size_t)16 * D_MODEL;
    const _Float16* bcol0 = Bt + (size_t)(n0 + nl) * D_MODEL;
    const _Float16* bcol1 = bcol0 + (size_t)16 * D_MODEL;

    v8f c00 = {}, c01 = {}, c10 = {}, c11 = {};
    for (int kk = 0; kk < D_MODEL; kk += 32) {
        v16h a0 = load_a_frag(arow0, kk, half);
        v16h a1 = load_a_frag(arow1, kk, half);
        v16h b0 = load_b_frag(bcol0, kk, half);
        v16h b1 = load_b_frag(bcol1, kk, half);
        c00 = __builtin_amdgcn_wmma_f32_16x16x32_f16(false, a0, false, b0, (short)0, c00, false, false);
        c01 = __builtin_amdgcn_wmma_f32_16x16x32_f16(false, a0, false, b1, (short)0, c01, false, false);
        c10 = __builtin_amdgcn_wmma_f32_16x16x32_f16(false, a1, false, b0, (short)0, c10, false, false);
        c11 = __builtin_amdgcn_wmma_f32_16x16x32_f16(false, a1, false, b1, (short)0, c11, false, false);
    }
    const float bv0 = bias[n0 + nl];
    const float bv1 = bias[n0 + 16 + nl];
    store_tile<MODE>(c00, r0,      n0,      bv0, out16, outf, half, nl);
    store_tile<MODE>(c01, r0,      n0 + 16, bv1, out16, outf, half, nl);
    store_tile<MODE>(c10, r0 + 16, n0,      bv0, out16, outf, half, nl);
    store_tile<MODE>(c11, r0 + 16, n0 + 16, bv1, out16, outf, half, nl);
}

// ---- fused flash attention: one wave = 16 query rows, online softmax,
// scores never leave the WGP. q16,k16: [B,H,S,64]; vT16: [B,H,64,S].
__global__ void attn_flash_wmma(const _Float16* __restrict__ q16,
                                const _Float16* __restrict__ k16,
                                const _Float16* __restrict__ vT16,
                                _Float16* __restrict__ ao16) {
    __shared__ _Float16 pbuf[8][16 * 32];   // per-wave P tile (16x32 f16)

    const int lane = threadIdx.x & 31;
    const int wave = threadIdx.x >> 5;
    const int half = lane >> 4;
    const int nl   = lane & 15;
    const int bh = blockIdx.y;              // 0..31 = b*16+h
    const int b_ = bh >> 4, h = bh & 15;
    const int q0 = blockIdx.x * 128 + wave * 16;

    const _Float16* qh = q16  + (size_t)bh * SEQ * D_K;
    const _Float16* kh = k16  + (size_t)bh * SEQ * D_K;
    const _Float16* vh = vT16 + (size_t)bh * D_K * SEQ;

    const _Float16* qrow = qh + (size_t)(q0 + nl) * D_K;
    const v16h aq0 = load_a_frag(qrow, 0,  half);   // d 0..31
    const v16h aq1 = load_a_frag(qrow, 32, half);   // d 32..63

    v8f o0 = {}, o1 = {}, o2 = {}, o3 = {};
    float mrow[8], lrow[8];
#pragma unroll
    for (int r = 0; r < 8; ++r) { mrow[r] = -1e30f; lrow[r] = 0.0f; }
    const float scale = 0.125f;             // 1/sqrt(d_k)

    for (int j0 = 0; j0 < SEQ; j0 += 32) {
        // --- scores: two 16x16 column tiles (keys j0+n, j0+16+n) ---
        const _Float16* kc0 = kh + (size_t)(j0 + nl) * D_K;
        const _Float16* kc1 = kh + (size_t)(j0 + 16 + nl) * D_K;
        v8f z = {};
        v16h bk = load_b_frag(kc0, 0, half);
        v8f c0 = __builtin_amdgcn_wmma_f32_16x16x32_f16(false, aq0, false, bk, (short)0, z, false, false);
        bk = load_b_frag(kc0, 32, half);
        c0 = __builtin_amdgcn_wmma_f32_16x16x32_f16(false, aq1, false, bk, (short)0, c0, false, false);
        bk = load_b_frag(kc1, 0, half);
        v8f c1 = __builtin_amdgcn_wmma_f32_16x16x32_f16(false, aq0, false, bk, (short)0, z, false, false);
        bk = load_b_frag(kc1, 32, half);
        c1 = __builtin_amdgcn_wmma_f32_16x16x32_f16(false, aq1, false, bk, (short)0, c1, false, false);

        // --- online softmax: VALU-only DPP16 row reductions ---
        float p0[8], p1[8], fac[8];
#pragma unroll
        for (int r = 0; r < 8; ++r) {
            float s0 = c0[r] * scale, s1 = c1[r] * scale;
            float tm = row16_max(fmaxf(s0, s1));
            float mn = fmaxf(mrow[r], tm);
            float f  = __expf(mrow[r] - mn);
            p0[r] = __expf(s0 - mn);
            p1[r] = __expf(s1 - mn);
            float rs = row16_sum(p0[r] + p1[r]);
            lrow[r] = lrow[r] * f + rs;
            mrow[r] = mn;
            fac[r]  = f;
        }
#pragma unroll
        for (int r = 0; r < 8; ++r) {
            o0[r] *= fac[r]; o1[r] *= fac[r]; o2[r] *= fac[r]; o3[r] *= fac[r];
        }

        // --- P (D-layout) -> LDS row-major -> A-fragment reshape ---
        _Float16* pw = pbuf[wave];
#pragma unroll
        for (int r = 0; r < 8; ++r) {
            const int m = r + 8 * half;
            pw[m * 32 + nl]      = (_Float16)p0[r];
            pw[m * 32 + 16 + nl] = (_Float16)p1[r];
        }
        asm volatile("s_wait_dscnt 0" ::: "memory");  // per-wave LDS is in-order; wait before re-read
        const v16h pa = load_a_frag(pw + nl * 32, 0, half);

        // --- O += P x V : 4 d-column tiles, K = 32 keys (contiguous in vT) ---
        v16h bv = load_b_frag(vh + (size_t)(0  + nl) * SEQ + j0, 0, half);
        o0 = __builtin_amdgcn_wmma_f32_16x16x32_f16(false, pa, false, bv, (short)0, o0, false, false);
        bv = load_b_frag(vh + (size_t)(16 + nl) * SEQ + j0, 0, half);
        o1 = __builtin_amdgcn_wmma_f32_16x16x32_f16(false, pa, false, bv, (short)0, o1, false, false);
        bv = load_b_frag(vh + (size_t)(32 + nl) * SEQ + j0, 0, half);
        o2 = __builtin_amdgcn_wmma_f32_16x16x32_f16(false, pa, false, bv, (short)0, o2, false, false);
        bv = load_b_frag(vh + (size_t)(48 + nl) * SEQ + j0, 0, half);
        o3 = __builtin_amdgcn_wmma_f32_16x16x32_f16(false, pa, false, bv, (short)0, o3, false, false);
    }

    // --- normalize and store attn output f16 in [B,S,D_MODEL] (heads concatenated) ---
#pragma unroll
    for (int r = 0; r < 8; ++r) {
        const float inv = 1.0f / lrow[r];
        const int m = r + 8 * half;
        const size_t base = ((size_t)(b_ * SEQ) + q0 + m) * D_MODEL + h * D_K;
        ao16[base + 0  + nl] = (_Float16)(o0[r] * inv);
        ao16[base + 16 + nl] = (_Float16)(o1[r] * inv);
        ao16[base + 32 + nl] = (_Float16)(o2[r] * inv);
        ao16[base + 48 + nl] = (_Float16)(o3[r] * inv);
    }
}

extern "C" void kernel_launch(void* const* d_in, const int* in_sizes, int n_in,
                              void* d_out, int out_size, void* d_ws, size_t ws_size,
                              hipStream_t stream) {
    const float* Q   = (const float*)d_in[0];
    const float* K   = (const float*)d_in[1];
    const float* V   = (const float*)d_in[2];
    const float* W_Q = (const float*)d_in[3];
    const float* b_Q = (const float*)d_in[4];
    const float* W_K = (const float*)d_in[5];
    const float* b_K = (const float*)d_in[6];
    const float* W_V = (const float*)d_in[7];
    const float* b_V = (const float*)d_in[8];
    const float* W_O = (const float*)d_in[9];
    const float* b_O = (const float*)d_in[10];
    float* out = (float*)d_out;

    const size_t NX = (size_t)ROWS * D_MODEL;     // 4,194,304 elems
    const size_t NW = (size_t)D_MODEL * D_MODEL;  // 1,048,576 elems
    _Float16* p   = (_Float16*)d_ws;
    _Float16* xq  = p;            p += NX;
    _Float16* xk  = p;            p += NX;
    _Float16* xv  = p;            p += NX;
    _Float16* wqT = p;            p += NW;
    _Float16* wkT = p;            p += NW;
    _Float16* wvT = p;            p += NW;
    _Float16* woT = p;            p += NW;
    _Float16* q16 = p;            p += NX;
    _Float16* k16 = p;            p += NX;
    _Float16* vT  = p;            p += NX;
    _Float16* ao  = p;            p += NX;   // total 64 MiB of f16 scratch

    // 1) convert activations to f16
    const int cvb = 256;
    const int cvg = (int)((NX + cvb - 1) / cvb);
    cvt_f32_to_f16<<<cvg, cvb, 0, stream>>>(Q, xq, (int)NX);
    cvt_f32_to_f16<<<cvg, cvb, 0, stream>>>(K, xk, (int)NX);
    cvt_f32_to_f16<<<cvg, cvb, 0, stream>>>(V, xv, (int)NX);

    // 2) transpose+convert weights
    dim3 tg(32, 32);
    transpose_w_f16<<<tg, 256, 0, stream>>>(W_Q, wqT);
    transpose_w_f16<<<tg, 256, 0, stream>>>(W_K, wkT);
    transpose_w_f16<<<tg, 256, 0, stream>>>(W_V, wvT);
    transpose_w_f16<<<tg, 256, 0, stream>>>(W_O, woT);

    // 3) projections (WMMA, 32x32 per wave): 4096x1024x1024 each
    dim3 gg(32, 16), gb(256);
    gemm_f16_wmma<0><<<gg, gb, 0, stream>>>(xq, wqT, b_Q, q16, nullptr);
    gemm_f16_wmma<0><<<gg, gb, 0, stream>>>(xk, wkT, b_K, k16, nullptr);
    gemm_f16_wmma<1><<<gg, gb, 0, stream>>>(xv, wvT, b_V, vT,  nullptr);

    // 4) fused flash attention (WMMA + DPP online softmax)
    attn_flash_wmma<<<dim3(16, 32), 256, 0, stream>>>(q16, k16, vT, ao);

    // 5) output projection -> f32 d_out
    gemm_f16_wmma<2><<<gg, gb, 0, stream>>>(ao, woT, b_O, nullptr, out);
}
